// CorrProductBlock_23424751632791
// MI455X (gfx1250) — matrix operand order repaired and checked by reading, not compiled, not run on hardware
//
#include <hip/hip_runtime.h>

// ---------------------------------------------------------------------------
// CorrProductBlock fused kernel for gfx1250 (MI455X, wave32, WMMA bf16).
//   out = concat( (p0@Wout0 + x0@Wsc0)*k , ((p1@Wout1 + x1@Wsc1)*k).reshape )
// All 16 of the 128x128 GEMMs run as V_WMMA_F32_16X16X32_BF16 with the 8
// weight matrices resident in LDS (bf16, pre-swizzled to B-fragment layout).
// Weight preload uses the Tensor Data Mover (TENSOR_LOAD_TO_LDS); next-tile
// node_feats are prefetched (global_prefetch_b8) to hide HBM latency.
// ---------------------------------------------------------------------------

#define MULC 128
#define N_NODES 100000
#define TILE 16            // nodes per workgroup tile (one WMMA M-tile)
#define BLOCK 256          // 8 waves of 32
#define FEAT 512           // 4*MUL (x0 | x1 regrouped component-major)
#define NMAT 8
#define MAT_ELEMS (MULC * MULC)          // 16384 bf16 per matrix
#define INV_SQRT_MUL 0.08838834764831845f
#define INV_SQRT_3   0.5773502691896258f

typedef __attribute__((ext_vector_type(16))) __bf16        v16bf;
typedef __attribute__((ext_vector_type(8)))  float         v8f;
typedef __attribute__((ext_vector_type(8)))  unsigned int  v8u;
typedef unsigned int u32x4 __attribute__((ext_vector_type(4)));
typedef int          i32x4 __attribute__((ext_vector_type(4)));
typedef int          i32x8 __attribute__((ext_vector_type(8)));

__device__ __forceinline__ unsigned short f2bf(float f) {
  unsigned int u = __float_as_uint(f);
  unsigned int r = u + 0x7FFFu + ((u >> 16) & 1u);   // round-to-nearest-even
  return (unsigned short)(r >> 16);
}
__device__ __forceinline__ float bf2f(unsigned short h) {
  return __uint_as_float(((unsigned int)h) << 16);
}
__device__ __forceinline__ v8f vzero8() {
  v8f z = {0.f, 0.f, 0.f, 0.f, 0.f, 0.f, 0.f, 0.f};
  return z;
}

// 16 contiguous bf16 (32 bytes, 16B-aligned) -> one WMMA fragment register set
__device__ __forceinline__ v16bf load_frag32(const unsigned short* p) {
  const uint4* q = (const uint4*)p;
  uint4 a = q[0], b = q[1];
  v8u u;
  u[0] = a.x; u[1] = a.y; u[2] = a.z; u[3] = a.w;
  u[4] = b.x; u[5] = b.y; u[6] = b.z; u[7] = b.w;
  return __builtin_bit_cast(v16bf, u);
}

// A fragment (16x32 bf16, M=16 nodes, K=32 channel slice) from an LDS
// activation buffer with row pitch FEAT.  ISA layout: lanes 0-15 hold row M
// with K = {ks*32 + 0..7, ks*32 + 16..23}; lanes 16-31 hold K+8.
__device__ __forceinline__ v16bf load_a_frag(const unsigned short* A, int lane,
                                             int featBase, int ks) {
  const int m  = lane & 15;
  const int kh = lane >> 4;
  const unsigned short* p = A + m * FEAT + featBase + ks * 32 + kh * 8;
  const uint4 lo = *(const uint4*)p;          // K offsets 0..7
  const uint4 hi = *(const uint4*)(p + 16);   // K offsets 16..23
  v8u u;
  u[0] = lo.x; u[1] = lo.y; u[2] = lo.z; u[3] = lo.w;
  u[4] = hi.x; u[5] = hi.y; u[6] = hi.z; u[7] = hi.w;
  return __builtin_bit_cast(v16bf, u);
}

// B fragment from the pre-swizzled weight image: block (ks, nt) is 32 lanes
// x 16 bf16, each lane's 16 values contiguous.
__device__ __forceinline__ v16bf load_b_frag(const unsigned short* Wlds,
                                             int mat, int ks, int nt, int lane) {
  return load_frag32(Wlds + mat * MAT_ELEMS + (ks * 8 + nt) * 512 + lane * 16);
}

__device__ __forceinline__ v8f wmma_bf16(v16bf a, v16bf b, v8f c) {
  return __builtin_amdgcn_wmma_f32_16x16x32_bf16(false, a, false, b,
                                                 (short)0, c, false, false);
}

// Scatter a scaled f32 accumulator tile into an LDS bf16 activation buffer.
// C layout: VGPR r -> M = r + 8*(lane>=16), N = lane&15.
__device__ __forceinline__ void store_acc_lds(unsigned short* Tb, v8f acc,
                                              int lane, int featBase, int n0,
                                              float scale) {
  const int kh  = lane >> 4;
  const int col = featBase + n0 + (lane & 15);
#pragma unroll
  for (int r = 0; r < 8; ++r) {
    Tb[(r + 8 * kh) * FEAT + col] = f2bf(acc[r] * scale);
  }
}

// ---------------------------------------------------------------------------
// Prep kernel: f32 128x128 row-major weights -> bf16, swizzled so every
// (ks, nt) B-fragment is 32 lanes x 16 contiguous bf16.
//   dst[mat*16384 + (ks*8+nt)*512 + lane*16 + j]
//     = bf16( W[mat][ (ks*32 + (j>>3)*16 + (lane>>4)*8 + (j&7)) * 128 + nt*16 + (lane&15) ] )
// ---------------------------------------------------------------------------
__global__ __launch_bounds__(256) void prep_weights(
    const float* __restrict__ W0, const float* __restrict__ W1,
    const float* __restrict__ W2, const float* __restrict__ W3,
    const float* __restrict__ W4, const float* __restrict__ W5,
    const float* __restrict__ W6, const float* __restrict__ W7,
    unsigned short* __restrict__ dst) {
  const int tid = blockIdx.x * blockDim.x + threadIdx.x;
  if (tid >= NMAT * MAT_ELEMS) return;
  const int mat = tid >> 14;
  const float* W;
  switch (mat) {
    case 0: W = W0; break; case 1: W = W1; break;
    case 2: W = W2; break; case 3: W = W3; break;
    case 4: W = W4; break; case 5: W = W5; break;
    case 6: W = W6; break; default: W = W7; break;
  }
  const int rem = tid & (MAT_ELEMS - 1);
  const int blk = rem >> 9;          // ks*8 + nt
  const int l   = (rem >> 4) & 31;   // lane
  const int j   = rem & 15;          // element within lane
  const int ks = blk >> 3, nt = blk & 7;
  const int row = ks * 32 + (j >> 3) * 16 + (l >> 4) * 8 + (j & 7);   // K index
  const int col = nt * 16 + (l & 15);            // N index
  dst[tid] = f2bf(W[row * MULC + col]);
}

// ---------------------------------------------------------------------------
// TDM preload of the 256KB swizzled-weight image into LDS.  One wave issues a
// single 2D descriptor (data_size=8B, tile 32768x1); tracked with TENSORcnt.
// D# bit packing per CDNA5 ISA ch.8 (group0 128b, group1 256b).
// ---------------------------------------------------------------------------
__device__ __forceinline__ void tdm_load_weights(unsigned short* Wlds,
                                                 const unsigned short* Wsw,
                                                 int tid) {
#if __has_builtin(__builtin_amdgcn_tensor_load_to_lds)
  if ((tid >> 5) == 0) {    // wave 0 only (EXEC ignored by tensor ops)
    const unsigned long long ga = (unsigned long long)(uintptr_t)Wsw;
    u32x4 g0;
    g0[0] = 1u;                                        // count=1, user D#
    g0[1] = (unsigned int)(uintptr_t)(void*)Wlds;      // lds_addr[31:0]
    g0[2] = (unsigned int)(ga & 0xFFFFFFFFu);          // global_addr[31:0]
    g0[3] = (unsigned int)((ga >> 32) & 0x1FFFFFFu)    // global_addr[56:32]
            | (2u << 30);                              // type=2 ("image")
    i32x8 g1;
    g1[0] = (int)(3u << 16);          // workgroup_mask=0, data_size=3 (8B)
    g1[1] = (int)(0x8000u << 16);     // tensor_dim0 = 32768 (low16 @ [63:48])
    g1[2] = (int)(1u << 16);          // tensor_dim0 hi=0; tensor_dim1 = 1
    g1[3] = (int)(0x8000u << 16);     // tensor_dim1 hi=0; tile_dim0 = 32768
    g1[4] = 1;                        // tile_dim1 = 1; tile_dim2 = 0
    g1[5] = 32768;                    // tensor_dim0_stride low32
    g1[6] = 0;                        // stride hi; tensor_dim1_stride lo
    g1[7] = 0;
    i32x4 z4 = {0, 0, 0, 0};          // groups 2/3 unused (<=2D tensor)
#if defined(__clang_major__) && (__clang_major__ >= 23)
    i32x8 z8 = {0, 0, 0, 0, 0, 0, 0, 0};
    __builtin_amdgcn_tensor_load_to_lds(g0, g1, z4, z4, z8, 0);
#else
    __builtin_amdgcn_tensor_load_to_lds(g0, g1, z4, z4, 0);
#endif
    __builtin_amdgcn_s_wait_tensorcnt(0);
  }
#else
  // Fallback: cooperative vector copy.
  const uint4* src = (const uint4*)Wsw;
  uint4* dstp = (uint4*)Wlds;
  for (int i = tid; i < NMAT * MAT_ELEMS / 8; i += BLOCK) dstp[i] = src[i];
#endif
}

// ---------------------------------------------------------------------------
// Fused main kernel. Persistent blocks grid-stride over 16-node tiles.
// LDS: 8 weight matrices (bf16, 256KB) + X/T/U activation buffers (3x16KB)
//      + 16 element ids  -> ~304KB of the 320KB WGP LDS.
// ---------------------------------------------------------------------------
__global__ __launch_bounds__(BLOCK) void corr_fused(
    const float* __restrict__ feats, const int* __restrict__ elems,
    const float* __restrict__ w1_0, const float* __restrict__ w1_1,
    const float* __restrict__ w2_00, const float* __restrict__ w2_11,
    const float* __restrict__ w2_01,
    const unsigned short* __restrict__ Wsw,
    float* __restrict__ out) {
  extern __shared__ unsigned short smem[];
  unsigned short* Wlds = smem;                        // 8*16384 bf16
  unsigned short* X    = Wlds + NMAT * MAT_ELEMS;     // staged input tile
  unsigned short* T    = X + TILE * FEAT;             // h -> a (in place)
  unsigned short* U    = T + TILE * FEAT;             // p
  int* elds = (int*)(U + TILE * FEAT);                // 16 element ids

  const int tid  = threadIdx.x;
  const int lane = tid & 31;
  const int wv   = tid >> 5;     // wave id == output n-tile (16 cols)
  const int n0   = wv * 16;

  // ---- TDM preload of all 8 swizzled weight matrices (once per block) ----
  tdm_load_weights(Wlds, Wsw, tid);
  __syncthreads();

  const int ntiles = N_NODES / TILE;   // 6250, exact
  for (int tile = blockIdx.x; tile < ntiles; tile += gridDim.x) {
    const int base = tile * TILE;

    // ---- stage node_feats tile into X (bf16, x1 regrouped comp-major) ----
    {
      const float4* src = (const float4*)(feats + (size_t)base * FEAT);
      for (int i = tid; i < TILE * FEAT / 4; i += BLOCK) {
        float4 v = src[i];
        const int n  = (i * 4) >> 9;     // node row (FEAT=512)
        const int sc0 = (i * 4) & 511;   // source column
#pragma unroll
        for (int q = 0; q < 4; ++q) {
          const int sc = sc0 + q;
          const float f = (q == 0) ? v.x : (q == 1) ? v.y : (q == 2) ? v.z : v.w;
          int dc;
          if (sc < MULC) dc = sc;                       // x0[c]
          else {                                        // x1[c,i] -> comp-major
            const int t2 = sc - MULC;
            dc = MULC + (t2 % 3) * MULC + (t2 / 3);
          }
          X[n * FEAT + dc] = f2bf(f);
        }
      }
      if (tid < TILE) elds[tid] = elems[base + tid];
    }
    // ---- prefetch next tile's node_feats (overlaps the 3 WMMA stages) ----
    {
      const int ntile = tile + gridDim.x;
      if (ntile < ntiles) {
        const char* np = (const char*)(feats + (size_t)ntile * TILE * FEAT);
        __builtin_prefetch(np + tid * 128, 0, 3);   // 256 thr * 128B = 32KB
      }
    }
    __syncthreads();

    // ---- stage 1: h0 = x0@Wpre0*k ; h1_i = x1_i@Wpre1*k  -> T ----
    {
      v16bf b[4];
#pragma unroll
      for (int ks = 0; ks < 4; ++ks) b[ks] = load_b_frag(Wlds, 0, ks, wv, lane);
      v8f acc = vzero8();
#pragma unroll
      for (int ks = 0; ks < 4; ++ks)
        acc = wmma_bf16(load_a_frag(X, lane, 0, ks), b[ks], acc);
      store_acc_lds(T, acc, lane, 0, n0, INV_SQRT_MUL);

#pragma unroll
      for (int ks = 0; ks < 4; ++ks) b[ks] = load_b_frag(Wlds, 1, ks, wv, lane);
#pragma unroll
      for (int comp = 0; comp < 3; ++comp) {
        const int fb = MULC + comp * MULC;
        v8f a1 = vzero8();
#pragma unroll
        for (int ks = 0; ks < 4; ++ks)
          a1 = wmma_bf16(load_a_frag(X, lane, fb, ks), b[ks], a1);
        store_acc_lds(T, a1, lane, fb, n0, INV_SQRT_MUL);
      }
    }
    __syncthreads();

    // ---- element-gated nonlinearity, per (node, channel), in place in T ----
#pragma unroll
    for (int t = 0; t < TILE * MULC / BLOCK; ++t) {   // 8 iters
      const int idx = tid + t * BLOCK;
      const int n = idx >> 7, c = idx & 127;
      const int wi = elds[n] * MULC + c;
      const float h0  = bf2f(T[n * FEAT + c]);
      const float h1x = bf2f(T[n * FEAT + MULC + c]);
      const float h1y = bf2f(T[n * FEAT + 2 * MULC + c]);
      const float h1z = bf2f(T[n * FEAT + 3 * MULC + c]);
      const float a0 = w1_0[wi] * h0 + w2_00[wi] * h0 * h0 +
                       w2_11[wi] * ((h1x * h1x + h1y * h1y + h1z * h1z) * INV_SQRT_3);
      const float g = w1_1[wi] + w2_01[wi] * h0;
      T[n * FEAT + c]            = f2bf(a0);
      T[n * FEAT + MULC + c]     = f2bf(g * h1x);
      T[n * FEAT + 2 * MULC + c] = f2bf(g * h1y);
      T[n * FEAT + 3 * MULC + c] = f2bf(g * h1z);
    }
    __syncthreads();

    // ---- stage 2: p0 = a0@Wprod0*k ; p1_i = a1_i@Wprod1*k  -> U ----
    {
      v16bf b[4];
#pragma unroll
      for (int ks = 0; ks < 4; ++ks) b[ks] = load_b_frag(Wlds, 2, ks, wv, lane);
      v8f acc = vzero8();
#pragma unroll
      for (int ks = 0; ks < 4; ++ks)
        acc = wmma_bf16(load_a_frag(T, lane, 0, ks), b[ks], acc);
      store_acc_lds(U, acc, lane, 0, n0, INV_SQRT_MUL);

#pragma unroll
      for (int ks = 0; ks < 4; ++ks) b[ks] = load_b_frag(Wlds, 3, ks, wv, lane);
#pragma unroll
      for (int comp = 0; comp < 3; ++comp) {
        const int fb = MULC + comp * MULC;
        v8f a1 = vzero8();
#pragma unroll
        for (int ks = 0; ks < 4; ++ks)
          a1 = wmma_bf16(load_a_frag(T, lane, fb, ks), b[ks], a1);
        store_acc_lds(U, a1, lane, fb, n0, INV_SQRT_MUL);
      }
    }
    __syncthreads();

    // ---- stage 3: u = (p@Wout + x@Wsc)*k  -> global (f32) ----
    {
      const int kh = lane >> 4;
      v16bf bo[4], bs[4];
      // scalar channel: Wout0 / Wsc0
#pragma unroll
      for (int ks = 0; ks < 4; ++ks) {
        bo[ks] = load_b_frag(Wlds, 4, ks, wv, lane);
        bs[ks] = load_b_frag(Wlds, 6, ks, wv, lane);
      }
      v8f acc = vzero8();
#pragma unroll
      for (int ks = 0; ks < 4; ++ks)
        acc = wmma_bf16(load_a_frag(U, lane, 0, ks), bo[ks], acc);
#pragma unroll
      for (int ks = 0; ks < 4; ++ks)
        acc = wmma_bf16(load_a_frag(X, lane, 0, ks), bs[ks], acc);
      {
        const int cc = n0 + (lane & 15);
#pragma unroll
        for (int r = 0; r < 8; ++r)
          out[(size_t)(base + r + 8 * kh) * FEAT + cc] = acc[r] * INV_SQRT_MUL;
      }
      // vector channels: Wout1 / Wsc1, interleaved output cols 128 + 3d + i
#pragma unroll
      for (int ks = 0; ks < 4; ++ks) {
        bo[ks] = load_b_frag(Wlds, 5, ks, wv, lane);
        bs[ks] = load_b_frag(Wlds, 7, ks, wv, lane);
      }
#pragma unroll
      for (int comp = 0; comp < 3; ++comp) {
        const int fb = MULC + comp * MULC;
        v8f a1 = vzero8();
#pragma unroll
        for (int ks = 0; ks < 4; ++ks)
          a1 = wmma_bf16(load_a_frag(U, lane, fb, ks), bo[ks], a1);
#pragma unroll
        for (int ks = 0; ks < 4; ++ks)
          a1 = wmma_bf16(load_a_frag(X, lane, fb, ks), bs[ks], a1);
        const int d  = n0 + (lane & 15);
        const int cc = MULC + 3 * d + comp;
#pragma unroll
        for (int r = 0; r < 8; ++r)
          out[(size_t)(base + r + 8 * kh) * FEAT + cc] = a1[r] * INV_SQRT_MUL;
      }
    }
    __syncthreads();   // X/T/U reused next tile
  }
}

// ---------------------------------------------------------------------------
extern "C" void kernel_launch(void* const* d_in, const int* in_sizes, int n_in,
                              void* d_out, int out_size, void* d_ws, size_t ws_size,
                              hipStream_t stream) {
  const float* feats = (const float*)d_in[0];
  const int*   elems = (const int*)d_in[1];
  // matrices in LDS slot order: Wpre0, Wpre1, Wprod0, Wprod1, Wout0, Wout1, Wsc0, Wsc1
  const float* Wpre0  = (const float*)d_in[2];
  const float* Wpre1  = (const float*)d_in[3];
  const float* w1_0   = (const float*)d_in[4];
  const float* w1_1   = (const float*)d_in[5];
  const float* w2_00  = (const float*)d_in[6];
  const float* w2_11  = (const float*)d_in[7];
  const float* w2_01  = (const float*)d_in[8];
  const float* Wprod0 = (const float*)d_in[9];
  const float* Wprod1 = (const float*)d_in[10];
  const float* Wout0  = (const float*)d_in[11];
  const float* Wout1  = (const float*)d_in[12];
  const float* Wsc0   = (const float*)d_in[13];
  const float* Wsc1   = (const float*)d_in[14];

  unsigned short* Wsw = (unsigned short*)d_ws;   // 8*16384 bf16 = 256KB

  prep_weights<<<(NMAT * MAT_ELEMS + 255) / 256, 256, 0, stream>>>(
      Wpre0, Wpre1, Wprod0, Wprod1, Wout0, Wout1, Wsc0, Wsc1, Wsw);

  const size_t smem_bytes =
      (size_t)(NMAT * MAT_ELEMS + 3 * TILE * FEAT) * sizeof(unsigned short) +
      TILE * sizeof(int);
  (void)hipFuncSetAttribute((const void*)corr_fused,
                            hipFuncAttributeMaxDynamicSharedMemorySize,
                            (int)smem_bytes);

  corr_fused<<<640, BLOCK, smem_bytes, stream>>>(
      feats, elems, w1_0, w1_1, w2_00, w2_11, w2_01, Wsw, (float*)d_out);
}
